// AttentionLSEncoder_75110388072500
// MI455X (gfx1250) — matrix operand order repaired and checked by reading, not compiled, not run on hardware
//
#include <hip/hip_runtime.h>
#include <cstdint>
#include <cstddef>

typedef __attribute__((ext_vector_type(16))) __bf16 v16bf;
typedef __attribute__((ext_vector_type(8)))  __bf16 v8bf;
typedef __attribute__((ext_vector_type(8)))  float  v8f;

#define DEV static __device__ __forceinline__

constexpr int B_   = 8;
constexpr int S_   = 2000;
constexpr int DM_  = 512;
constexpr int H_   = 4;
constexpr int DK_  = 512;
constexpr int HDK_ = 2048;   // H*DK
constexpr int NL_  = 32;
constexpr int WS_  = 10;
constexpr int EXT_ = 5;
constexpr int T_   = 20;     // 2*EXT + WS
constexpr int G_   = 200;    // S/WS
constexpr int SPAD_= 2016;   // S padded to multiple of 32 (63*32)
constexpr float NEGINF = -1.0e30f;

// ---------------- WMMA fragment helpers ----------------
// A (16x32 bf16): lane L holds row M=L%16; half=L/16:
//   elems 0..7  = K in [half*16,  half*16+8)
//   elems 8..15 = K in [half*16+8, half*16+16)
DEV v16bf afrag_load(const __bf16* rowptr, int k0, int half) {
  v16bf a;
  v8bf x0 = *(const v8bf*)(rowptr + k0 + half * 16);
  v8bf x1 = *(const v8bf*)(rowptr + k0 + half * 16 + 8);
#pragma unroll
  for (int j = 0; j < 8; ++j) { a[j] = x0[j]; a[j + 8] = x1[j]; }
  return a;
}

DEV v16bf frag_zero() {
  v16bf a;
#pragma unroll
  for (int j = 0; j < 16; ++j) a[j] = (__bf16)0.0f;
  return a;
}

// B (32x16 bf16): lane L holds column N=L%16; half=L/16:
//   elems j = K row (half*16 + j). p points at the 16 contiguous values.
DEV v16bf bfrag_packed(const __bf16* p) {
  v16bf bfr;
  v8bf x0 = *(const v8bf*)(p);
  v8bf x1 = *(const v8bf*)(p + 8);
#pragma unroll
  for (int j = 0; j < 8; ++j) { bfr[j] = x0[j]; bfr[j + 8] = x1[j]; }
  return bfr;
}

DEV v8f wmma_bf16(v16bf a, v16bf b, v8f c) {
  return __builtin_amdgcn_wmma_f32_16x16x32_bf16(
      /*neg_a=*/false, a, /*neg_b=*/false, b,
      /*c_mod=*/(short)0, c, /*reuse_a=*/false, /*reuse_b=*/false);
}

// ---------------- elementwise / packing ----------------
__global__ void f32_to_bf16(const float* __restrict__ in, __bf16* __restrict__ out, size_t n) {
  size_t i = (size_t)blockIdx.x * blockDim.x + threadIdx.x;
  size_t stride = (size_t)gridDim.x * blockDim.x;
  for (; i < n; i += stride) out[i] = (__bf16)in[i];
}

// W [Kd,N] f32 row-major -> Wp bf16 [Kd/32][N][32]:
//   Wp[((kb*N)+c)*32 + rr] = W[(kb*32+rr)*N + c]
__global__ void pack_w(const float* __restrict__ W, __bf16* __restrict__ Wp, int Kd, int N) {
  size_t total = (size_t)Kd * N;
  size_t i = (size_t)blockIdx.x * blockDim.x + threadIdx.x;
  size_t stride = (size_t)gridDim.x * blockDim.x;
  for (; i < total; i += stride) {
    int rr = (int)(i & 31);
    int c  = (int)((i >> 5) % (size_t)N);
    int kb = (int)(i / ((size_t)32 * N));
    Wp[i] = (__bf16)W[(size_t)(kb * 32 + rr) * N + c];
  }
}

// ---------------- main WMMA GEMM: out = A[M,Kd](bf16) x Wp(packed) + bias ----------------
// block = 256 thr = 8 waves; tile 128 rows x 64 cols; wave w -> rows 16w..16w+15, all 64 cols.
// B tile (packed, 64 cols x 32 k = 4KB) staged through double-buffered LDS once per block;
// A fragments register-pipelined one k-step ahead; B prefetched two k-steps ahead.
__global__ __launch_bounds__(256) void gemm_xw(
    const __bf16* __restrict__ A, const __bf16* __restrict__ Wp,
    const float* __restrict__ bias, float* __restrict__ outF,
    __bf16* __restrict__ outB, int M, int N, int Kd, float scale, int write_bf16)
{
  const int tid  = threadIdx.x;
  const int wave = tid >> 5, lane = tid & 31;
  const int half = lane >> 4, lcol = lane & 15;
  const int rbase = blockIdx.x * 128 + wave * 16;
  const int cbase = blockIdx.y * 64;
  const int row = rbase + lcol;
  const __bf16* arow = A + (size_t)row * Kd;

  __shared__ __align__(16) __bf16 Bs[2][64 * 32];   // 2 x 4KB double buffer

  v8f acc[4];
#pragma unroll
  for (int nt = 0; nt < 4; ++nt)
#pragma unroll
    for (int v = 0; v < 8; ++v) acc[nt][v] = 0.0f;

  const int ksteps = Kd >> 5;

  // one coalesced b128 per thread stages the whole 64x32 packed B tile of step kb
  auto stageB = [&](int kb, int buf) {
    const __bf16* src = Wp + ((size_t)(kb * N + cbase)) * 32 + tid * 8;
    *(v8bf*)(&Bs[buf][tid * 8]) = *(const v8bf*)(src);
  };

  // prologue: stage k-step 0, pipeline A for k-step 0
  stageB(0, 0);
  v16bf a_cur = afrag_load(arow, 0, half);
  __syncthreads();

  int buf = 0;
  for (int kb = 0; kb < ksteps; ++kb) {
    v16bf a_nxt = a_cur;
    const int nkb = kb + 1;
    if (nkb < ksteps) {
      stageB(nkb, buf ^ 1);                    // overlaps with WMMAs below
      a_nxt = afrag_load(arow, nkb * 32, half);
    }
    if (kb + 2 < ksteps) {                     // gfx1250 global_prefetch_b8
      const __bf16* pf = Wp + ((size_t)((kb + 2) * N + cbase)) * 32 + tid * 8;
      __builtin_prefetch(pf, 0, 1);
    }
#pragma unroll
    for (int nt = 0; nt < 4; ++nt) {
      const __bf16* bp = &Bs[buf][(nt * 16 + lcol) * 32 + half * 16];
      v16bf b = bfrag_packed(bp);
      acc[nt] = wmma_bf16(a_cur, b, acc[nt]);
    }
    __syncthreads();                           // stage(nkb) visible, buf reusable
    a_cur = a_nxt;
    buf ^= 1;
  }

#pragma unroll
  for (int nt = 0; nt < 4; ++nt)
#pragma unroll
    for (int v = 0; v < 8; ++v) {
      int m = rbase + v + 8 * half;
      int n = cbase + nt * 16 + lcol;
      float val = (acc[nt][v] + bias[n]) * scale;
      if (write_bf16) outB[(size_t)m * N + n] = (__bf16)val;
      else            outF[(size_t)m * N + n] = val;
    }
}

// ---------------- row LayerNorm over HDK, f32 in -> bf16 out ----------------
__global__ __launch_bounds__(256) void ln_rows(
    const float* __restrict__ in, const float* __restrict__ gg,
    const float* __restrict__ bb, __bf16* __restrict__ out)
{
  const int row = blockIdx.x, tid = threadIdx.x;
  __shared__ float r1[256], r2[256];
  const float* x = in + (size_t)row * HDK_;
  float s = 0.f, sq = 0.f;
  for (int c = tid; c < HDK_; c += 256) { float v = x[c]; s += v; sq += v * v; }
  r1[tid] = s; r2[tid] = sq; __syncthreads();
  for (int o = 128; o > 0; o >>= 1) {
    if (tid < o) { r1[tid] += r1[tid + o]; r2[tid] += r2[tid + o]; }
    __syncthreads();
  }
  float mean = r1[0] * (1.0f / HDK_);
  float var  = r2[0] * (1.0f / HDK_) - mean * mean;
  float rstd = rsqrtf(var + 1e-5f);
  for (int c = tid; c < HDK_; c += 256)
    out[(size_t)row * HDK_ + c] = (__bf16)((x[c] - mean) * rstd * gg[c] + bb[c]);
}

// ---------------- softmax of hs over S (axis=1), write transposed+padded bf16 ----------------
// hs [B,S,H*NL] f32 -> hsT [B,H,NL,SPAD] bf16
__global__ __launch_bounds__(256) void softmax_s(
    const float* __restrict__ hs, const float* __restrict__ mask,
    __bf16* __restrict__ hsT)
{
  const int b = blockIdx.x >> 7;     // 128 cols per batch
  const int j = blockIdx.x & 127;
  const int h = j / NL_, l = j % NL_;
  const int tid = threadIdx.x;
  __shared__ float red[256];

  float mx = NEGINF;
  for (int s = tid; s < S_; s += 256)
    if (mask[(size_t)b * S_ + s] != 0.0f)
      mx = fmaxf(mx, hs[((size_t)b * S_ + s) * (H_ * NL_) + j]);
  red[tid] = mx; __syncthreads();
  for (int o = 128; o > 0; o >>= 1) { if (tid < o) red[tid] = fmaxf(red[tid], red[tid + o]); __syncthreads(); }
  mx = red[0]; __syncthreads();

  float sum = 0.f;
  for (int s = tid; s < S_; s += 256)
    if (mask[(size_t)b * S_ + s] != 0.0f)
      sum += expf(hs[((size_t)b * S_ + s) * (H_ * NL_) + j] - mx);
  red[tid] = sum; __syncthreads();
  for (int o = 128; o > 0; o >>= 1) { if (tid < o) red[tid] += red[tid + o]; __syncthreads(); }
  float inv = (red[0] > 0.f) ? 1.0f / red[0] : 0.f;

  __bf16* dst = hsT + ((size_t)(b * H_ + h) * NL_ + l) * SPAD_;
  for (int s = tid; s < SPAD_; s += 256) {
    float p = 0.f;
    if (s < S_ && mask[(size_t)b * S_ + s] != 0.0f)
      p = expf(hs[((size_t)b * S_ + s) * (H_ * NL_) + j] - mx) * inv;
    dst[s] = (__bf16)p;
  }
}

// ---------------- landmark compression: outC[b,l,h*DK+d] = sum_s hsT[b,h,l,s]*KV[b,s,h*DK+d] ----
// grid.x = b*H+h (32), grid.y = 64-col chunk of DK (8); 128 thr = 4 waves.
__global__ __launch_bounds__(128) void gemm_kc(
    const __bf16* __restrict__ hsT, const __bf16* __restrict__ KVb,
    float* __restrict__ outC)
{
  const int bh = blockIdx.x, b = bh / H_, h = bh % H_;
  const int tid = threadIdx.x, wave = tid >> 5, lane = tid & 31;
  const int half = lane >> 4, lcol = lane & 15;
  const int rbase = (wave >> 1) * 16;                       // 0 or 16
  const int cbase = blockIdx.y * 64 + (wave & 1) * 32;      // 32-col strip per wave
  const __bf16* arow = hsT + ((size_t)(b * H_ + h) * NL_ + rbase + lcol) * SPAD_;

  v8f acc[2];
#pragma unroll
  for (int nt = 0; nt < 2; ++nt)
#pragma unroll
    for (int v = 0; v < 8; ++v) acc[nt][v] = 0.0f;

  for (int kb = 0; kb < SPAD_ / 32; ++kb) {
    v16bf a = afrag_load(arow, kb * 32, half);
#pragma unroll
    for (int nt = 0; nt < 2; ++nt) {
      int d = cbase + nt * 16 + lcol;
      v16bf bfr;
#pragma unroll
      for (int j = 0; j < 16; ++j) {
        int sk = kb * 32 + half * 16 + j;
        bfr[j] = (sk < S_) ? KVb[((size_t)b * S_ + sk) * HDK_ + h * DK_ + d]
                           : (__bf16)0.0f;
      }
      acc[nt] = wmma_bf16(a, bfr, acc[nt]);
    }
  }
#pragma unroll
  for (int nt = 0; nt < 2; ++nt)
#pragma unroll
    for (int v = 0; v < 8; ++v) {
      int l = rbase + v + 8 * half;
      int d = cbase + nt * 16 + lcol;
      outC[((size_t)b * NL_ + l) * HDK_ + h * DK_ + d] = acc[nt][v];
    }
}

// ---------------- fused attention per (b,h,g): scores -> softmax -> context ----------------
__global__ __launch_bounds__(128) void attn_fused(
    const __bf16* __restrict__ Qb, const __bf16* __restrict__ Kb,
    const __bf16* __restrict__ Vb, const __bf16* __restrict__ Kcb,
    const __bf16* __restrict__ Vcb, const float* __restrict__ mask,
    __bf16* __restrict__ Cb)
{
  const int blk = blockIdx.x;
  const int g = blk % G_;
  const int h = (blk / G_) % H_;
  const int b = blk / (G_ * H_);
  const int tid = threadIdx.x, wave = tid >> 5, lane = tid & 31;
  const int half = lane >> 4, lcol = lane & 15;

  __shared__ float sc[16][64];
  __shared__ __align__(16) __bf16 pr[16][64];

  // ---- phase 1: one 16x16 score tile per wave (cols: 0-31 landmarks, 32-63 window) ----
  {
    v8f acc;
#pragma unroll
    for (int v = 0; v < 8; ++v) acc[v] = 0.0f;

    const int sq = g * WS_ + lcol;
    const bool qok = (sq < S_);
    const __bf16* qrow = Qb + ((size_t)b * S_ + (qok ? sq : 0)) * HDK_ + h * DK_;

    const __bf16* bptr = nullptr;
    bool bok = false;
    if (wave < 2) {
      int l = wave * 16 + lcol;
      bptr = Kcb + ((size_t)b * NL_ + l) * HDK_ + h * DK_;
      bok = true;
    } else {
      int t = (wave - 2) * 16 + lcol;
      int sk = g * WS_ - EXT_ + t;
      if (t < T_ && sk >= 0 && sk < S_) {
        bptr = Kb + ((size_t)b * S_ + sk) * HDK_ + h * DK_;
        bok = true;
      }
    }
    for (int kb = 0; kb < DK_ / 32; ++kb) {
      v16bf a  = qok ? afrag_load(qrow, kb * 32, half) : frag_zero();
      v16bf bf = bok ? bfrag_packed(bptr + kb * 32 + half * 16) : frag_zero();
      acc = wmma_bf16(a, bf, acc);
    }
#pragma unroll
    for (int v = 0; v < 8; ++v) {
      int m = v + 8 * half;
      int col = wave * 16 + lcol;
      float val = acc[v];
      if (col >= NL_) {                      // window part: mask invalid keys
        int t = col - NL_;
        int sk = g * WS_ - EXT_ + t;
        if (t >= T_ || sk < 0 || sk >= S_ || mask[(size_t)b * S_ + sk] == 0.0f)
          val = NEGINF;
      }
      sc[m][col] = val;
    }
  }
  __syncthreads();

  // ---- phase 2: softmax over the 52 (padded to 64) score columns ----
  if (tid < 16) {
    int m = tid;
    int sq = g * WS_ + m;
    bool valid = (m < WS_) && (sq < S_) && (mask[(size_t)b * S_ + sq] != 0.0f);
    if (valid) {
      float mx = NEGINF;
      for (int c = 0; c < 64; ++c) mx = fmaxf(mx, sc[m][c]);
      float sum = 0.f;
      for (int c = 0; c < 64; ++c) sum += expf(sc[m][c] - mx);
      float inv = (sum > 0.f) ? 1.0f / sum : 0.f;
      for (int c = 0; c < 64; ++c) pr[m][c] = (__bf16)(expf(sc[m][c] - mx) * inv);
    } else {
      for (int c = 0; c < 64; ++c) pr[m][c] = (__bf16)0.0f;
    }
  }
  __syncthreads();

  // ---- phase 3: context tile [16 x 512] = P[16x64] x [V_c ; V_tiles][64x512] ----
  {
    const int dbase = wave * 128;    // each wave: 128 of the 512 dims (8 n-tiles)
    v8f acc[8];
#pragma unroll
    for (int nt = 0; nt < 8; ++nt)
#pragma unroll
      for (int v = 0; v < 8; ++v) acc[nt][v] = 0.0f;

    for (int kb2 = 0; kb2 < 2; ++kb2) {
      v16bf a = afrag_load(&pr[lcol][0], kb2 * 32, half);
#pragma unroll
      for (int nt = 0; nt < 8; ++nt) {
        int d = dbase + nt * 16 + lcol;
        v16bf bf;
#pragma unroll
        for (int j = 0; j < 16; ++j) {
          int k = kb2 * 32 + half * 16 + j;
          __bf16 val = (__bf16)0.0f;
          if (k < NL_) {
            val = Vcb[((size_t)b * NL_ + k) * HDK_ + h * DK_ + d];
          } else if (k < NL_ + T_) {
            int sk = g * WS_ - EXT_ + (k - NL_);
            if (sk >= 0 && sk < S_)
              val = Vb[((size_t)b * S_ + sk) * HDK_ + h * DK_ + d];
          }
          bf[j] = val;
        }
        acc[nt] = wmma_bf16(a, bf, acc[nt]);
      }
    }
#pragma unroll
    for (int nt = 0; nt < 8; ++nt)
#pragma unroll
      for (int v = 0; v < 8; ++v) {
        int m = v + 8 * half;
        int sq = g * WS_ + m;
        if (m < WS_ && sq < S_) {
          int d = dbase + nt * 16 + lcol;
          Cb[((size_t)b * S_ + sq) * HDK_ + h * DK_ + d] = (__bf16)acc[nt][v];
        }
      }
  }
}

// ---------------- host: launch sequence ----------------
extern "C" void kernel_launch(void* const* d_in, const int* in_sizes, int n_in,
                              void* d_out, int out_size, void* d_ws, size_t ws_size,
                              hipStream_t stream) {
  (void)in_sizes; (void)n_in; (void)out_size; (void)ws_size;
  const float* X     = (const float*)d_in[0];
  const float* mask  = (const float*)d_in[3];
  const float* Wq    = (const float*)d_in[4];
  const float* bq    = (const float*)d_in[5];
  const float* Wk    = (const float*)d_in[6];
  const float* bk    = (const float*)d_in[7];
  const float* Wv    = (const float*)d_in[8];
  const float* bv    = (const float*)d_in[9];
  const float* Wo    = (const float*)d_in[10];
  const float* bo    = (const float*)d_in[11];
  const float* Wd    = (const float*)d_in[12];
  const float* bd    = (const float*)d_in[13];
  const float* lnl_g = (const float*)d_in[14];
  const float* lnl_b = (const float*)d_in[15];
  const float* lns_g = (const float*)d_in[16];
  const float* lns_b = (const float*)d_in[17];

  char* ws = (char*)d_ws;
  size_t off = 0;
  auto carve = [&](size_t bytes) -> char* {
    char* p = ws + off;
    off += (bytes + 255) & ~(size_t)255;
    return p;
  };
  const int M = B_ * S_;   // 16000
  __bf16* Xb  = (__bf16*)carve((size_t)M * DM_ * 2);
  __bf16* Wqp = (__bf16*)carve((size_t)DM_ * HDK_ * 2);
  __bf16* Wkp = (__bf16*)carve((size_t)DM_ * HDK_ * 2);
  __bf16* Wvp = (__bf16*)carve((size_t)DM_ * HDK_ * 2);
  __bf16* Wdp = (__bf16*)carve((size_t)DM_ * (H_ * NL_) * 2);
  __bf16* Wop = (__bf16*)carve((size_t)HDK_ * DM_ * 2);
  __bf16* Qb  = (__bf16*)carve((size_t)M * HDK_ * 2);
  __bf16* Kb  = (__bf16*)carve((size_t)M * HDK_ * 2);
  __bf16* Vb  = (__bf16*)carve((size_t)M * HDK_ * 2);
  __bf16* Cb  = (__bf16*)carve((size_t)M * HDK_ * 2);
  float*  tmpF= (float* )carve((size_t)M * HDK_ * 4);
  float*  hsF = (float* )carve((size_t)M * (H_ * NL_) * 4);
  __bf16* hsT = (__bf16*)carve((size_t)B_ * H_ * NL_ * SPAD_ * 2);
  float*  KcF = (float* )carve((size_t)B_ * NL_ * HDK_ * 4);
  float*  VcF = (float* )carve((size_t)B_ * NL_ * HDK_ * 4);
  __bf16* Kcb = (__bf16*)carve((size_t)B_ * NL_ * HDK_ * 2);
  __bf16* Vcb = (__bf16*)carve((size_t)B_ * NL_ * HDK_ * 2);

  const float qscale = 0.04419417382415922f;  // 1/sqrt(DK)

  // bf16 conversion + weight packing
  f32_to_bf16<<<2048, 256, 0, stream>>>(X, Xb, (size_t)M * DM_);
  pack_w<<<1024, 256, 0, stream>>>(Wq, Wqp, DM_, HDK_);
  pack_w<<<1024, 256, 0, stream>>>(Wk, Wkp, DM_, HDK_);
  pack_w<<<1024, 256, 0, stream>>>(Wv, Wvp, DM_, HDK_);
  pack_w<<< 256, 256, 0, stream>>>(Wd, Wdp, DM_, H_ * NL_);
  pack_w<<<1024, 256, 0, stream>>>(Wo, Wop, HDK_, DM_);

  // projections
  dim3 gproj(M / 128, HDK_ / 64);
  gemm_xw<<<gproj, 256, 0, stream>>>(Xb, Wqp, bq, nullptr, Qb, M, HDK_, DM_, qscale, 1);
  gemm_xw<<<gproj, 256, 0, stream>>>(Xb, Wkp, bk, tmpF, nullptr, M, HDK_, DM_, 1.0f, 0);
  ln_rows<<<M, 256, 0, stream>>>(tmpF, lnl_g, lnl_b, Kb);
  gemm_xw<<<gproj, 256, 0, stream>>>(Xb, Wvp, bv, tmpF, nullptr, M, HDK_, DM_, 1.0f, 0);
  ln_rows<<<M, 256, 0, stream>>>(tmpF, lnl_g, lnl_b, Vb);

  // landmark weights + softmax over S
  dim3 ghs(M / 128, (H_ * NL_) / 64);
  gemm_xw<<<ghs, 256, 0, stream>>>(Xb, Wdp, bd, hsF, nullptr, M, H_ * NL_, DM_, 1.0f, 0);
  softmax_s<<<B_ * H_ * NL_, 256, 0, stream>>>(hsF, mask, hsT);

  // compressed K/V + LN
  dim3 gkc(B_ * H_, DK_ / 64);
  gemm_kc<<<gkc, 128, 0, stream>>>(hsT, Kb, KcF);
  gemm_kc<<<gkc, 128, 0, stream>>>(hsT, Vb, VcF);
  ln_rows<<<B_ * NL_, 256, 0, stream>>>(KcF, lns_g, lns_b, Kcb);
  ln_rows<<<B_ * NL_, 256, 0, stream>>>(VcF, lns_g, lns_b, Vcb);

  // fused attention (compressed + window) -> Cb
  attn_fused<<<B_ * H_ * G_, 128, 0, stream>>>(Qb, Kb, Vb, Kcb, Vcb, mask, Cb);

  // output projection -> f32 d_out
  dim3 gout(M / 128, DM_ / 64);
  gemm_xw<<<gout, 256, 0, stream>>>(Cb, Wop, bo, (float*)d_out, nullptr, M, DM_, HDK_, 1.0f, 0);
}
